// MambaLanguageModel_36687610642862
// MI455X (gfx1250) — compile-verified, matrix operand order checked by reading, not run on hardware
//
#include <hip/hip_runtime.h>
#include <hip/hip_bf16.h>

// ---------------- model constants ----------------
#define VOCAB 32000
#define DM    768
#define NLAY  4
#define NST   16      // d_state
#define KCONV 4
#define ED    1536    // d_inner
#define DTR   48      // dt_rank
#define DTP   64      // dt_rank padded to K-multiple of 32
#define BATCH 2
#define SEQL  2048
#define BL    (BATCH*SEQL)

#define MT 4          // 16-row tiles per wave  (64 rows)

typedef __attribute__((ext_vector_type(16))) __bf16 v16bf;
typedef __attribute__((ext_vector_type(8)))  float  v8f;

union BfFrag {
    v16bf    v;
    unsigned p[8];
};

// Pack two f32 into one dword of two bf16 (truncation) with a single v_perm_b32.
// dst bytes = { hi[3], hi[2], lo[3], lo[2] }  -> selector 0x07060302
__device__ __forceinline__ unsigned pack_bf(float lo, float hi) {
    return __builtin_amdgcn_perm(__float_as_uint(hi), __float_as_uint(lo),
                                 0x07060302u);
}

// Load one 16x32 bf16 fragment for this lane.
// p = row base + 8*half; chunks at +0 (K 0..7 rel) and +16 (K 16..23 rel).
// Matches ISA 7.12.2 "16-bit A-Matrix 16x32" VGPR layout.
__device__ __forceinline__ void load_frag(const float* __restrict__ p, BfFrag& f) {
    const float4 x0 = ((const float4*)p)[0];
    const float4 x1 = ((const float4*)p)[1];
    const float4 x2 = ((const float4*)(p + 16))[0];
    const float4 x3 = ((const float4*)(p + 16))[1];
    f.p[0] = pack_bf(x0.x, x0.y);
    f.p[1] = pack_bf(x0.z, x0.w);
    f.p[2] = pack_bf(x1.x, x1.y);
    f.p[3] = pack_bf(x1.z, x1.w);
    f.p[4] = pack_bf(x2.x, x2.y);
    f.p[5] = pack_bf(x2.z, x2.w);
    f.p[6] = pack_bf(x3.x, x3.y);
    f.p[7] = pack_bf(x3.z, x3.w);
}

// ---------------------------------------------------------------------------
// WMMA GEMM: C[M,N] = A[M,K] * W[N,K]^T (+ bias[n]) (+ softplus)
// One wave per 64 x (NTILES*16) macro-tile.
// Requires M % 64 == 0, K % 32 == 0, N % (NTILES*16) == 0.
// No conditionals in the hot loop -> EXEC stays all-1s for every WMMA.
// ---------------------------------------------------------------------------
template <int NTILES>
__global__ __launch_bounds__(128)
void gemm_wmma_bt(const float* __restrict__ A, int lda,
                  const float* __restrict__ W, int ldw,
                  float* __restrict__ C, int ldc,
                  int M, int N, int K,
                  const float* __restrict__ bias, int act)
{
    const int lane = threadIdx.x & 31;
    const int wave = threadIdx.x >> 5;
    const int nmac = N / (NTILES * 16);
    const int mmac = M / (MT * 16);
    const int mac  = blockIdx.x * 4 + wave;
    if (mac >= mmac * nmac) return;                 // wave-uniform exit

    const int tm = mac / nmac;
    const int tn = mac - tm * nmac;
    const int m0 = tm * (MT * 16);
    const int n0 = tn * (NTILES * 16);
    const int half = lane >> 4;
    const int r    = lane & 15;

    // 32-bit element offsets off uniform bases (all buffers < 4GB)
    const unsigned aoff = (unsigned)(m0 + r) * (unsigned)lda + 8u * half;
    const unsigned woff = (unsigned)(n0 + r) * (unsigned)ldw + 8u * half;

    v8f acc[MT][NTILES];
    #pragma unroll
    for (int i = 0; i < MT; ++i)
        #pragma unroll
        for (int j = 0; j < NTILES; ++j)
            acc[i][j] = (v8f){0.f, 0.f, 0.f, 0.f, 0.f, 0.f, 0.f, 0.f};

    for (int k0 = 0; k0 < K; k0 += 32) {
        BfFrag b[NTILES];
        #pragma unroll
        for (int j = 0; j < NTILES; ++j) {
            const float* wp = W + woff + (unsigned)(j * 16) * (unsigned)ldw + k0;
            load_frag(wp, b[j]);
            __builtin_prefetch(wp + 32, 0, 1);
        }
        #pragma unroll
        for (int i = 0; i < MT; ++i) {
            BfFrag a;
            load_frag(A + aoff + (unsigned)(i * 16) * (unsigned)lda + k0, a);
            #pragma unroll
            for (int j = 0; j < NTILES; ++j) {
                acc[i][j] = __builtin_amdgcn_wmma_f32_16x16x32_bf16(
                    false, a.v, false, b[j].v, (short)0, acc[i][j],
                    false, false);
            }
        }
    }

    #pragma unroll
    for (int j = 0; j < NTILES; ++j) {
        const int n = n0 + j * 16 + r;
        const float bv = bias ? bias[n] : 0.f;
        #pragma unroll
        for (int i = 0; i < MT; ++i) {
            #pragma unroll
            for (int v = 0; v < 8; ++v) {
                const int m = m0 + i * 16 + v + 8 * half;
                float val = acc[i][j][v] + bv;
                if (act == 1)
                    val = (val > 20.f) ? val : log1pf(expf(val));
                C[(size_t)m * ldc + n] = val;
            }
        }
    }
}

// ---------------------------------------------------------------------------
// Embedding gather: x[m, d] = emb[ids[m], d]
// ---------------------------------------------------------------------------
__global__ __launch_bounds__(256)
void embed_k(const int* __restrict__ ids, const float* __restrict__ emb,
             float* __restrict__ x, int total)
{
    int idx = blockIdx.x * blockDim.x + threadIdx.x;
    if (idx >= total) return;
    int row = idx / DM;
    int d   = idx - row * DM;
    x[idx] = emb[(size_t)ids[row] * DM + d];
}

// ---------------------------------------------------------------------------
// Depthwise causal conv (K=4) + bias + SiLU.  u-part is cols [0,ED) of xz.
// ---------------------------------------------------------------------------
__global__ __launch_bounds__(256)
void conv_silu_k(const float* __restrict__ xz, const float* __restrict__ cw,
                 const float* __restrict__ cb, float* __restrict__ u)
{
    int idx = blockIdx.x * blockDim.x + threadIdx.x;
    if (idx >= BL * ED) return;
    int e = idx % ED;
    int g = idx / ED;           // b*L + l
    int l = g % SEQL;
    int b = g / SEQL;

    float s = cb[e];
    #pragma unroll
    for (int j = 0; j < KCONV; ++j) {
        int ls = l - (KCONV - 1) + j;
        if (ls >= 0)
            s += cw[e * KCONV + j] * xz[(size_t)(b * SEQL + ls) * (2 * ED) + e];
    }
    u[idx] = s / (1.f + expf(-s));   // silu
}

// ---------------------------------------------------------------------------
// Zero-pad dt activations: dbc[:, 0:48] -> dtp (BL x 64)
// ---------------------------------------------------------------------------
__global__ __launch_bounds__(256)
void pad_dt_act_k(const float* __restrict__ dbc, float* __restrict__ dtp)
{
    int idx = blockIdx.x * blockDim.x + threadIdx.x;
    if (idx >= BL * DTP) return;
    int col = idx % DTP;
    int row = idx / DTP;
    dtp[idx] = (col < DTR) ? dbc[(size_t)row * (DTR + 2 * NST) + col] : 0.f;
}

// Zero-pad dt weights: w_dt (ED x 48) -> wp (ED x 64)
__global__ __launch_bounds__(256)
void pad_dt_w_k(const float* __restrict__ w_dt, float* __restrict__ wp)
{
    int idx = blockIdx.x * blockDim.x + threadIdx.x;
    if (idx >= ED * DTP) return;
    int col = idx % DTP;
    int row = idx / DTP;
    wp[idx] = (col < DTR) ? w_dt[(size_t)row * DTR + col] : 0.f;
}

// ---------------------------------------------------------------------------
// Selective scan: one thread per (b, e) channel. 16-wide state in registers.
// Fuses  y = scan + u*D,  y *= silu(res).  Writes y in-place over u.
// ---------------------------------------------------------------------------
__global__ __launch_bounds__(256)
void scan_k(const float* __restrict__ delta, const float* __restrict__ dbc,
            const float* __restrict__ xz, const float* __restrict__ a_log,
            const float* __restrict__ Dp, float* __restrict__ u)
{
    int t = blockIdx.x * blockDim.x + threadIdx.x;
    if (t >= BATCH * ED) return;
    int e = t % ED;
    int b = t / ED;

    float Aa[NST], h[NST];
    #pragma unroll
    for (int n = 0; n < NST; ++n) {
        Aa[n] = -expf(a_log[(size_t)e * NST + n]);
        h[n]  = 0.f;
    }
    const float Dv = Dp[e];

    for (int l = 0; l < SEQL; ++l) {
        const size_t g  = (size_t)(b * SEQL + l);
        const float  d  = delta[g * ED + e];
        const float  uu = u[g * ED + e];
        const float  du = d * uu;
        const float* bc = dbc + g * (DTR + 2 * NST) + DTR;

        float y = 0.f;
        #pragma unroll
        for (int n = 0; n < NST; ++n) {
            h[n] = expf(d * Aa[n]) * h[n] + du * bc[n];
            y   += h[n] * bc[NST + n];
        }
        const float rv  = xz[g * (2 * ED) + ED + e];
        const float out = (y + uu * Dv) * (rv / (1.f + expf(-rv)));
        u[g * ED + e] = out;
    }
}

// ---------------------------------------------------------------------------
// Row LayerNorm (in-place): 1 block per row of DM=768.
// ---------------------------------------------------------------------------
__global__ __launch_bounds__(256)
void ln_k(float* __restrict__ x, const float* __restrict__ g,
          const float* __restrict__ bta)
{
    __shared__ float red[256];
    float* xr = x + (size_t)blockIdx.x * DM;

    float s = 0.f;
    for (int d = threadIdx.x; d < DM; d += 256) s += xr[d];
    red[threadIdx.x] = s;
    __syncthreads();
    for (int o = 128; o > 0; o >>= 1) {
        if (threadIdx.x < o) red[threadIdx.x] += red[threadIdx.x + o];
        __syncthreads();
    }
    const float mu = red[0] / DM;
    __syncthreads();

    float v = 0.f;
    for (int d = threadIdx.x; d < DM; d += 256) {
        float tdv = xr[d] - mu;
        v += tdv * tdv;
    }
    red[threadIdx.x] = v;
    __syncthreads();
    for (int o = 128; o > 0; o >>= 1) {
        if (threadIdx.x < o) red[threadIdx.x] += red[threadIdx.x + o];
        __syncthreads();
    }
    const float rs = rsqrtf(red[0] / DM + 1e-5f);
    __syncthreads();

    for (int d = threadIdx.x; d < DM; d += 256)
        xr[d] = (xr[d] - mu) * rs * g[d] + bta[d];
}

// ---------------------------------------------------------------------------
extern "C" void kernel_launch(void* const* d_in, const int* in_sizes, int n_in,
                              void* d_out, int out_size, void* d_ws, size_t ws_size,
                              hipStream_t stream)
{
    const int*   ids       = (const int*)  d_in[0];
    const float* emb       = (const float*)d_in[1];
    const float* in_proj_w = (const float*)d_in[2];
    const float* conv_w    = (const float*)d_in[3];
    const float* conv_b    = (const float*)d_in[4];
    const float* x_proj_w  = (const float*)d_in[5];
    const float* dt_proj_w = (const float*)d_in[6];
    const float* dt_proj_b = (const float*)d_in[7];
    const float* A_log     = (const float*)d_in[8];
    const float* Dp        = (const float*)d_in[9];
    const float* out_proj_w= (const float*)d_in[10];
    const float* ln_g      = (const float*)d_in[11];
    const float* ln_b      = (const float*)d_in[12];
    const float* head_w    = (const float*)d_in[13];
    const float* head_b    = (const float*)d_in[14];
    float*       out       = (float*)d_out;

    // ---- workspace allocation (all 256B aligned) ----
    char*  ws  = (char*)d_ws;
    size_t off = 0;
    auto alloc = [&](size_t bytes) -> float* {
        float* p = (float*)(ws + off);
        off = (off + bytes + 255) & ~(size_t)255;
        return p;
    };
    float* xbuf  = alloc((size_t)BL * DM * 4);              // (BL, 768)
    float* xz    = alloc((size_t)BL * 2 * ED * 4);          // (BL, 3072)
    float* u     = alloc((size_t)BL * ED * 4);              // (BL, 1536) -> y
    float* dbc   = alloc((size_t)BL * (DTR + 2 * NST) * 4); // (BL, 80)
    float* delta = alloc((size_t)BL * ED * 4);              // (BL, 1536)
    float* dtp   = alloc((size_t)BL * DTP * 4);             // (BL, 64) padded dt
    float* wdtp  = alloc((size_t)ED * DTP * 4);             // (1536, 64) padded W
    (void)ws_size; (void)in_sizes; (void)n_in; (void)out_size;

    // N % 32 == 0 path (2 n-tiles per wave)
    auto gemm2 = [&](const float* A, int lda, const float* W, int ldw,
                     float* C, int ldc, int M, int N, int K,
                     const float* bias, int act) {
        int tiles  = (M / 64) * (N / 32);
        int blocks = (tiles + 3) / 4;
        gemm_wmma_bt<2><<<blocks, 128, 0, stream>>>(A, lda, W, ldw, C, ldc,
                                                    M, N, K, bias, act);
    };
    // N % 16 == 0 path (1 n-tile per wave) -- used for ragged N=80
    auto gemm1 = [&](const float* A, int lda, const float* W, int ldw,
                     float* C, int ldc, int M, int N, int K,
                     const float* bias, int act) {
        int tiles  = (M / 64) * (N / 16);
        int blocks = (tiles + 3) / 4;
        gemm_wmma_bt<1><<<blocks, 128, 0, stream>>>(A, lda, W, ldw, C, ldc,
                                                    M, N, K, bias, act);
    };

    // 1) embedding
    {
        int total = BL * DM;
        embed_k<<<(total + 255) / 256, 256, 0, stream>>>(ids, emb, xbuf, total);
    }

    // 2) mamba layers
    for (int i = 0; i < NLAY; ++i) {
        const float* w_in  = in_proj_w  + (size_t)i * (2 * ED) * DM;
        const float* cw    = conv_w     + (size_t)i * ED * KCONV;
        const float* cb    = conv_b     + (size_t)i * ED;
        const float* w_x   = x_proj_w   + (size_t)i * (DTR + 2 * NST) * ED;
        const float* w_dt  = dt_proj_w  + (size_t)i * ED * DTR;
        const float* b_dt  = dt_proj_b  + (size_t)i * ED;
        const float* al    = A_log      + (size_t)i * ED * NST;
        const float* dpar  = Dp         + (size_t)i * ED;
        const float* w_out = out_proj_w + (size_t)i * DM * ED;

        // xz = x @ w_in^T          (4096 x 3072, K=768)
        gemm2(xbuf, DM, w_in, DM, xz, 2 * ED, BL, 2 * ED, DM, nullptr, 0);

        // u = silu(causal_dwconv(xz[:, :ED]) + cb)
        {
            int total = BL * ED;
            conv_silu_k<<<(total + 255) / 256, 256, 0, stream>>>(xz, cw, cb, u);
        }

        // dbc = u @ w_x^T          (4096 x 80, K=1536)  -- ragged N, NT=1
        gemm1(u, ED, w_x, ED, dbc, DTR + 2 * NST, BL, DTR + 2 * NST, ED,
              nullptr, 0);

        // pad dt activations and dt weights to K=64
        {
            int ta = BL * DTP;
            pad_dt_act_k<<<(ta + 255) / 256, 256, 0, stream>>>(dbc, dtp);
            int tw = ED * DTP;
            pad_dt_w_k<<<(tw + 255) / 256, 256, 0, stream>>>(w_dt, wdtp);
        }

        // delta = softplus(dtp @ wdtp^T + b_dt)   (4096 x 1536, K=64)
        gemm2(dtp, DTP, wdtp, DTP, delta, ED, BL, ED, DTP, b_dt, 1);

        // selective scan (writes gated y in-place over u)
        {
            int total = BATCH * ED;
            scan_k<<<(total + 255) / 256, 256, 0, stream>>>(delta, dbc, xz,
                                                            al, dpar, u);
        }

        // x = y @ w_out^T          (4096 x 768, K=1536)
        gemm2(u, ED, w_out, ED, xbuf, DM, BL, DM, ED, nullptr, 0);
    }

    // 3) layernorm (in-place on xbuf)
    ln_k<<<BL, 256, 0, stream>>>(xbuf, ln_g, ln_b);

    // 4) LM head: out = x @ head_w^T + head_b   (4096 x 32000, K=768)
    gemm2(xbuf, DM, head_w, DM, out, VOCAB, BL, VOCAB, DM, head_b, 0);
}
